// DynGRU_29746943492385
// MI455X (gfx1250) — matrix operand order repaired and checked by reading, not compiled, not run on hardware
//
#include <hip/hip_runtime.h>
#include <math.h>

typedef __attribute__((ext_vector_type(16))) __bf16 v16bf;
typedef __attribute__((ext_vector_type(8)))  float  v8f;
typedef __attribute__((ext_vector_type(4)))  unsigned int v4u;
typedef __attribute__((ext_vector_type(8)))  int v8i32;
typedef __attribute__((ext_vector_type(4)))  int v4i32;

#define TOK  32768   // B*L
#define BB   32
#define LL   1024
#define DIN  64
#define DM   512
#define DOUT 64
#define DH   512
#define N3M  1536
#define RBLK 16      // recurrence blocks (one WGP each)
#define PADW 520     // 512 elems + 8-elem (16 B) TDM pad -> conflict-free DS reads
#define CPOL_DEV 0x10  // gfx12 CPol: scope=DEV (cross-WGP visibility for state)

__device__ __forceinline__ unsigned short f2bf(float f) {
  union { float f; unsigned u; } x; x.f = f;
  unsigned u = x.u;
  return (unsigned short)((u + 0x7FFFu + ((u >> 16) & 1u)) >> 16);
}
__device__ __forceinline__ float bf2f(unsigned short b) {
  union { unsigned u; float f; } x; x.u = ((unsigned)b) << 16;
  return x.f;
}

// ---------------- Tensor Data Mover helpers (ISA 08_async_tensor §8) ----------
template <int CPOL>
__device__ __forceinline__ void tdm_issue_load(v4u g0, v8i32 g1, v4i32 g2, v4i32 g3) {
#if defined(__clang_major__) && (__clang_major__ >= 23)
  v8i32 z8 = {0, 0, 0, 0, 0, 0, 0, 0};
  __builtin_amdgcn_tensor_load_to_lds(g0, g1, g2, g3, z8, CPOL);
#else
  __builtin_amdgcn_tensor_load_to_lds(g0, g1, g2, g3, CPOL);
#endif
}
template <int CPOL>
__device__ __forceinline__ void tdm_issue_store(v4u g0, v8i32 g1, v4i32 g2, v4i32 g3) {
#if defined(__clang_major__) && (__clang_major__ >= 23)
  v8i32 z8 = {0, 0, 0, 0, 0, 0, 0, 0};
  __builtin_amdgcn_tensor_store_from_lds(g0, g1, g2, g3, z8, CPOL);
#else
  __builtin_amdgcn_tensor_store_from_lds(g0, g1, g2, g3, CPOL);
#endif
}

__device__ __forceinline__ v4u tdm_g0(unsigned lds_off, const void* gp) {
  unsigned long long ga = (unsigned long long)(size_t)gp;
  v4u g0;
  g0[0] = 1u;                                   // count=1, user descriptor
  g0[1] = lds_off;                              // lds_addr (bytes)
  g0[2] = (unsigned)ga;                         // global_addr[31:0]
  g0[3] = (unsigned)((ga >> 32) & 0x01FFFFFFull) | (2u << 30);  // addr[56:32]|type=2
  return g0;
}

// contiguous 1-D copy, 8-byte elements; optional LDS row padding
// ((pad_amt+1) dwords inserted every (1<<pad_int) 8-byte units -> bank-conflict-free rows)
template <int CPOL>
__device__ __forceinline__ void tdm_load_1d(unsigned lds_off, const void* gp,
                                            unsigned bytes, unsigned pad_int,
                                            unsigned pad_amt) {
  v4u g0 = tdm_g0(lds_off, gp);
  const unsigned n8 = bytes >> 3;
  unsigned w0 = (3u << 16);                      // data_size = 8 B
  if (pad_amt) w0 |= (1u << 20) | (pad_int << 22) | (pad_amt << 25);
  v8i32 g1;
  g1[0] = (int)w0;
  g1[1] = (int)((n8 & 0xFFFFu) << 16);           // tensor_dim0[15:0]
  g1[2] = (int)((n8 >> 16) & 0xFFFFu);           // tensor_dim0[31:16], dim1=0
  g1[3] = (int)((n8 & 0xFFFFu) << 16);           // tile_dim0
  g1[4] = 0;                                     // 1-D
  g1[5] = (int)n8;                               // stride0 (unused)
  g1[6] = 0; g1[7] = 0;
  v4i32 z = {0, 0, 0, 0};
  tdm_issue_load<CPOL>(g0, g1, z, z);
}

// 3-D gate tile: x = 32 contiguous cols, y = 3 chunks (stride 512),
// z = 32 batch rows (stride L*3M). 2-byte elems. LDS order = [row][chunk][col].
template <int CPOL>
__device__ __forceinline__ void tdm_load_gates(unsigned lds_off, const void* gp) {
  v4u g0 = tdm_g0(lds_off, gp);
  const unsigned long long s1 = (unsigned long long)LL * N3M;
  v8i32 g1;
  g1[0] = (int)(1u << 16);                       // data_size=2B
  g1[1] = (int)(32u << 16);                      // tensor_dim0 = 32
  g1[2] = (int)(3u << 16);                       // tensor_dim1 = 3
  g1[3] = (int)(32u << 16);                      // tile_dim0 = 32
  g1[4] = (int)(3u | (32u << 16));               // tile_dim1 = 3, tile_dim2 = 32
  g1[5] = (int)512;                              // stride0 = 512 (chunk spacing)
  g1[6] = (int)((s1 & 0xFFFFull) << 16);         // stride1[15:0]
  g1[7] = (int)((s1 >> 16) & 0xFFFFFFFFull);     // stride1[47:16]
  v4i32 g2 = {32, 0, 0, 0};                      // tensor_dim2 = 32
  v4i32 z = {0, 0, 0, 0};
  tdm_issue_load<CPOL>(g0, g1, g2, z);
}

// 2-D store: cols contiguous 2-byte elems, rows strided; LDS source is compact
template <int CPOL>
__device__ __forceinline__ void tdm_store_2d(unsigned lds_off, void* gp,
                                             unsigned cols, unsigned rows,
                                             unsigned long long stride0) {
  v4u g0 = tdm_g0(lds_off, gp);
  v8i32 g1;
  g1[0] = (int)(1u << 16);                       // data_size=2B
  g1[1] = (int)((cols & 0xFFFFu) << 16);         // tensor_dim0
  g1[2] = (int)((rows & 0xFFFFu) << 16);         // tensor_dim1
  g1[3] = (int)((cols & 0xFFFFu) << 16);         // tile_dim0
  g1[4] = (int)rows;                             // tile_dim1
  g1[5] = (int)(unsigned)stride0;                // stride0 lo32
  g1[6] = (int)((stride0 >> 32) & 0xFFFFull);    // stride0 hi16
  g1[7] = 0;
  v4i32 z = {0, 0, 0, 0};
  tdm_issue_store<CPOL>(g0, g1, z, z);
}
// ------------------------------------------------------------------------------

__global__ void cvt_f32_bf16(const float* __restrict__ src,
                             unsigned short* __restrict__ dst, int n) {
  int i = blockIdx.x * blockDim.x + threadIdx.x;
  if (i < n) dst[i] = f2bf(src[i]);
}

__global__ void zero_u32(unsigned int* __restrict__ p, int n) {
  int i = blockIdx.x * blockDim.x + threadIdx.x;
  if (i < n) p[i] = 0u;
}

// out[T,N] = act(A[T,K](bf16) @ W[N,K](bf16)^T + bias)
// The 16 x K A-tile is TDM-staged (row-padded) into LDS and shared by 8 waves;
// each wave computes a 16x64 strip (4 N-tiles).
template <int ACT, int OUTF>  // ACT: 1=relu  OUTF: 0=bf16 out, 1=f32 out
__global__ void wmma_gemm(const unsigned short* __restrict__ A,
                          const unsigned short* __restrict__ W,
                          const float* __restrict__ bias,
                          void* __restrict__ out,
                          int T, int N, int K) {
  (void)T;
  __shared__ unsigned short sA[16 * PADW];  // padded rows, K <= 512
  const int lane = threadIdx.x & 31;
  const int wave = threadIdx.x >> 5;
  const int row0 = blockIdx.x * 16;
  const int nBase = (blockIdx.y * (blockDim.x >> 5) + wave) * 64;
  const int m = lane & 15;
  const int g = lane >> 4;
  const int kpad = K + 8;

  if (wave == 0) {
    const unsigned pi = (K >= 512) ? 7u : (K >= 256) ? 6u : (K >= 128) ? 5u : 4u;
    tdm_load_1d<0>((unsigned)(size_t)sA, A + (size_t)row0 * K,
                   16u * (unsigned)K * 2u, pi, 3u);
    __builtin_amdgcn_s_wait_tensorcnt(0);
  }
  __syncthreads();

  if (nBase < N) {
    v8f acc[4] = {};
    for (int kb = 0; kb < K; kb += 32) {
      union { v16bf v; unsigned u[8]; } af;
      const unsigned short* arow = sA + m * kpad + kb;
#pragma unroll
      for (int v = 0; v < 8; ++v) {
        const int ko = (v < 4 ? 2 * v : 16 + 2 * (v - 4)) + 8 * g;
        af.u[v] = *(const unsigned*)(arow + ko);
      }
#pragma unroll
      for (int nt = 0; nt < 4; ++nt) {
        union { v16bf v; unsigned u[8]; } bw;
        const int n = nBase + nt * 16 + m;
        const unsigned short* wr = W + (size_t)n * K + kb;
#pragma unroll
        for (int v = 0; v < 8; ++v) bw.u[v] = *(const unsigned*)(wr + (v < 4 ? 2 * v : 16 + 2 * (v - 4)) + 8 * g);
        acc[nt] = __builtin_amdgcn_wmma_f32_16x16x32_bf16(
            false, af.v, false, bw.v, (short)0, acc[nt], false, false);
      }
    }
#pragma unroll
    for (int nt = 0; nt < 4; ++nt) {
      const int n = nBase + nt * 16 + m;
      const float bv = bias[n];
#pragma unroll
      for (int v = 0; v < 8; ++v) {
        float x = acc[nt][v] + bv;
        if (ACT) x = fmaxf(x, 0.0f);
        const size_t rr = (size_t)(row0 + v + 8 * g);
        if (OUTF == 0) ((unsigned short*)out)[rr * N + n] = f2bf(x);
        else           ((float*)out)[rr * N + n] = x;
      }
    }
  }
}

__device__ __forceinline__ void grid_barrier(unsigned int* ctr, unsigned int target) {
  __syncthreads();
  if (threadIdx.x == 0) {
    __threadfence();
    __hip_atomic_fetch_add(ctr, 1u, __ATOMIC_ACQ_REL, __HIP_MEMORY_SCOPE_AGENT);
    while (__hip_atomic_load(ctr, __ATOMIC_ACQUIRE, __HIP_MEMORY_SCOPE_AGENT) < target)
      __builtin_amdgcn_s_sleep(1);
  }
  __syncthreads();
}

// Persistent scan: 16 blocks x 192 threads (6 waves). Block b owns gate columns
// [32b, 32b+32). Its 96 mW rows live in LDS for the whole scan (one-time TDM,
// row-padded). Per step: TDM stages H (padded) + the 3-chunk gate tile, waves run
// pure LDS-fed WMMA, gating is block-local in LDS, and h_next leaves via two 2-D
// TENSOR_STORE_FROM_LDS descriptors. One agent-scope barrier per step.
__global__ void recurrence_kernel(const unsigned short* __restrict__ xenc, // [TOK,1536]
                                  const unsigned short* __restrict__ mWb,  // [1536,512]
                                  const float* __restrict__ mb,            // [1536]
                                  unsigned short* __restrict__ h0b,        // ping
                                  unsigned short* __restrict__ h1b,        // pong
                                  unsigned short* __restrict__ hseq,       // [TOK,512]
                                  unsigned int* __restrict__ ctr) {
  __shared__ unsigned short sW[3 * 32 * PADW];  // 97.5 KB resident weights
  __shared__ unsigned short sH[32 * PADW];      // 32.5 KB state tile (padded)
  __shared__ unsigned short sX[32 * 3 * 32];    //  6 KB gate inputs [row][chunk][col]
  __shared__ float hp[3 * 32 * 32];             // 12 KB hp partials
  __shared__ unsigned short sHn[32 * 32];       //  2 KB h_next (bf16, store staging)
  __shared__ float sF[32 * 32];                 //  4 KB f32 master state (block-local)
  const int tid = threadIdx.x;
  const int lane = tid & 31, wave = tid >> 5;
  const int b = blockIdx.x;
  const int chunk = wave >> 1;                  // 0..2 -> a / c / h
  const int ntile = wave & 1;                   // 0..1
  const int colBase = chunk * DM + b * 32 + ntile * 16;
  const int m = lane & 15, g = lane >> 4;
  const unsigned sH_off = (unsigned)(size_t)sH;
  const unsigned sX_off = (unsigned)(size_t)sX;
  const unsigned sHn_off = (unsigned)(size_t)sHn;

  if (wave == 0) {
    const unsigned sW_off = (unsigned)(size_t)sW;
    for (int c = 0; c < 3; ++c)  // 3 contiguous 32-row slabs of mW
      tdm_load_1d<0>(sW_off + (unsigned)(c * 32 * PADW * 2),
                     mWb + ((size_t)c * DM + (size_t)b * 32) * DM,
                     32u * DM * 2u, 7u, 3u);
    __builtin_amdgcn_s_wait_tensorcnt(0);
  }
  for (int idx = b * blockDim.x + tid; idx < BB * DM; idx += RBLK * blockDim.x)
    h0b[idx] = 0;
  for (int idx = tid; idx < 32 * 32; idx += blockDim.x) sF[idx] = 0.0f;
  unsigned int bar = 1;
  grid_barrier(ctr, bar * RBLK);

  const unsigned short* wl = sW + (chunk * 32 + ntile * 16 + m) * PADW;

  for (int t = 0; t < LL; ++t) {
    const unsigned short* hR = (t & 1) ? h1b : h0b;
    unsigned short*       hW = (t & 1) ? h0b : h1b;

    if (wave == 0) {
      tdm_load_1d<CPOL_DEV>(sH_off, hR, BB * DM * 2u, 7u, 3u);
      tdm_load_gates<0>(sX_off, xenc + (size_t)t * N3M + (size_t)b * 32);
      __builtin_amdgcn_s_wait_tensorcnt(0);
    }
    __syncthreads();

    v8f acc0 = {}, acc1 = {};
#pragma unroll 4
    for (int kb = 0; kb < DM; kb += 32) {
      union { v16bf v; unsigned u[8]; } a0, a1, bw;
      const unsigned short* ar0 = sH + m * PADW + kb;
      const unsigned short* ar1 = sH + (16 + m) * PADW + kb;
#pragma unroll
      for (int v = 0; v < 8; ++v) {
        const int ko = (v < 4 ? 2 * v : 16 + 2 * (v - 4)) + 8 * g;
        a0.u[v] = *(const unsigned*)(ar0 + ko);
        a1.u[v] = *(const unsigned*)(ar1 + ko);
      }
      const unsigned short* wr = wl + kb;
#pragma unroll
      for (int v = 0; v < 8; ++v) bw.u[v] = *(const unsigned*)(wr + (v < 4 ? 2 * v : 16 + 2 * (v - 4)) + 8 * g);
      acc0 = __builtin_amdgcn_wmma_f32_16x16x32_bf16(false, a0.v, false, bw.v,
                                                     (short)0, acc0, false, false);
      acc1 = __builtin_amdgcn_wmma_f32_16x16x32_bf16(false, a1.v, false, bw.v,
                                                     (short)0, acc1, false, false);
    }
    // deposit hp (+bias) into LDS: rows 0..15 from acc0, 16..31 from acc1
    const int cl = ntile * 16 + m;
    const float bv = mb[colBase + m];
#pragma unroll
    for (int v = 0; v < 8; ++v) {
      hp[(chunk * 32 + v + 8 * g) * 32 + cl]      = acc0[v] + bv;
      hp[(chunk * 32 + 16 + v + 8 * g) * 32 + cl] = acc1[v] + bv;
    }
    __syncthreads();

    // gating for this block's 32x32 slice (everything in LDS)
    for (int idx = tid; idx < 32 * 32; idx += blockDim.x) {
      const int i = idx >> 5;
      const int jl = idx & 31;
      const float xa = bf2f(sX[(i * 3 + 0) * 32 + jl]);
      const float xc = bf2f(sX[(i * 3 + 1) * 32 + jl]);
      const float xh = bf2f(sX[(i * 3 + 2) * 32 + jl]);
      const float ha = hp[(0 * 32 + i) * 32 + jl];
      const float hc = hp[(1 * 32 + i) * 32 + jl];
      const float hh = hp[(2 * 32 + i) * 32 + jl];
      const float av = 1.0f / (1.0f + __expf(-(xa + ha)));
      const float cv = 1.0f / (1.0f + __expf(-(xc + hc)));
      const float hold = sF[idx];
      const float hn = (1.0f - cv) * hold + cv * tanhf(xh + av * hh);
      sF[idx] = hn;
      sHn[idx] = f2bf(hn);
    }
    __syncthreads();

    if (wave == 0) {  // DMA h_next out: ping-pong state + h_seq column slab
      tdm_store_2d<CPOL_DEV>(sHn_off, hW + (size_t)b * 32, 32u, 32u,
                             (unsigned long long)DM);
      tdm_store_2d<CPOL_DEV>(sHn_off, hseq + (size_t)t * DM + (size_t)b * 32, 32u, 32u,
                             (unsigned long long)LL * DM);
      __builtin_amdgcn_s_wait_tensorcnt(0);
    }
    ++bar;
    grid_barrier(ctr, bar * RBLK);
  }
}

extern "C" void kernel_launch(void* const* d_in, const int* in_sizes, int n_in,
                              void* d_out, int out_size, void* d_ws, size_t ws_size,
                              hipStream_t stream) {
  (void)in_sizes; (void)n_in; (void)out_size; (void)ws_size;
  const float* x_seq = (const float*)d_in[0];
  const float* sW0 = (const float*)d_in[1];
  const float* sb0 = (const float*)d_in[2];
  const float* sW1 = (const float*)d_in[3];
  const float* sb1 = (const float*)d_in[4];
  const float* sWt = (const float*)d_in[5];
  const float* sbt = (const float*)d_in[6];
  const float* aW0 = (const float*)d_in[7];
  const float* ab0 = (const float*)d_in[8];
  const float* aW1 = (const float*)d_in[9];
  const float* ab1 = (const float*)d_in[10];
  const float* aWt = (const float*)d_in[11];
  const float* abt = (const float*)d_in[12];
  const float* mW  = (const float*)d_in[13];
  const float* mb  = (const float*)d_in[14];

  char* ws = (char*)d_ws;
  size_t off = 0;
  auto alloc = [&](size_t bytes) -> char* {
    char* p = ws + off;
    off += (bytes + 255) & ~(size_t)255;
    return p;
  };
  unsigned short* xb   = (unsigned short*)alloc((size_t)TOK * DIN * 2);
  unsigned short* sW0b = (unsigned short*)alloc((size_t)DH * DIN * 2);
  unsigned short* sW1b = (unsigned short*)alloc((size_t)DH * DH * 2);
  unsigned short* sWtb = (unsigned short*)alloc((size_t)N3M * DH * 2);
  unsigned short* aW0b = (unsigned short*)alloc((size_t)DH * DM * 2);
  unsigned short* aW1b = (unsigned short*)alloc((size_t)DH * DH * 2);
  unsigned short* aWtb = (unsigned short*)alloc((size_t)DOUT * DH * 2);
  unsigned short* mWb  = (unsigned short*)alloc((size_t)N3M * DM * 2);
  unsigned short* act1 = (unsigned short*)alloc((size_t)TOK * DH * 2);
  unsigned short* act2 = (unsigned short*)alloc((size_t)TOK * DH * 2);
  unsigned short* xenc = (unsigned short*)alloc((size_t)TOK * N3M * 2);
  unsigned short* hseq = (unsigned short*)alloc((size_t)TOK * DM * 2);
  unsigned short* h0b  = (unsigned short*)alloc((size_t)BB * DM * 2);
  unsigned short* h1b  = (unsigned short*)alloc((size_t)BB * DM * 2);
  unsigned int*   ctr  = (unsigned int*)alloc(256);

  auto cvt = [&](const float* s, unsigned short* d, int n) {
    cvt_f32_bf16<<<dim3((n + 255) / 256), dim3(256), 0, stream>>>(s, d, n);
  };
  cvt(x_seq, xb, TOK * DIN);
  cvt(sW0, sW0b, DH * DIN);
  cvt(sW1, sW1b, DH * DH);
  cvt(sWt, sWtb, N3M * DH);
  cvt(aW0, aW0b, DH * DM);
  cvt(aW1, aW1b, DH * DH);
  cvt(aWt, aWtb, DOUT * DH);
  cvt(mW, mWb, N3M * DM);

  // sensor MLP -> x_enc
  {
    dim3 grid(TOK / 16, 1);
    wmma_gemm<1, 0><<<grid, 256, 0, stream>>>(xb, sW0b, sb0, act1, TOK, DH, DIN);
    wmma_gemm<1, 0><<<grid, 256, 0, stream>>>(act1, sW1b, sb1, act2, TOK, DH, DH);
    dim3 grid3(TOK / 16, 3);
    wmma_gemm<0, 0><<<grid3, 256, 0, stream>>>(act2, sWtb, sbt, xenc, TOK, N3M, DH);
  }

  // sequential scan (persistent kernel, one grid barrier per step)
  zero_u32<<<1, 64, 0, stream>>>(ctr, 64);
  recurrence_kernel<<<dim3(RBLK), dim3(192), 0, stream>>>(xenc, mWb, mb, h0b, h1b,
                                                          hseq, ctr);

  // actuator MLP -> d_out (f32)
  {
    dim3 grid(TOK / 16, 1);
    wmma_gemm<1, 0><<<grid, 256, 0, stream>>>(hseq, aW0b, ab0, act1, TOK, DH, DM);
    wmma_gemm<1, 0><<<grid, 256, 0, stream>>>(act1, aW1b, ab1, act2, TOK, DH, DH);
    wmma_gemm<0, 1><<<grid, 256, 0, stream>>>(act2, aWtb, abt, d_out, TOK, DOUT, DH);
  }
}
